// Qwen3_5LinearDecoderLayer_31791347925074
// MI455X (gfx1250) — matrix operand order repaired and checked by reading, not compiled, not run on hardware
//
#include <hip/hip_runtime.h>

// ---------------- types for WMMA ----------------
typedef __attribute__((ext_vector_type(8)))  float        v8f;
typedef __attribute__((ext_vector_type(16))) __bf16       v16bf;
typedef __attribute__((ext_vector_type(8)))  unsigned int u32x8;

union Frag { u32x8 u; v16bf b; };

__device__ __forceinline__ unsigned short f2bf(float f) {
  unsigned int u = __float_as_uint(f);
  u += 0x7FFFu + ((u >> 16) & 1u);           // round-to-nearest-even
  return (unsigned short)(u >> 16);
}
__device__ __forceinline__ unsigned int pack2bf(float a, float b) {
  return (unsigned int)f2bf(a) | ((unsigned int)f2bf(b) << 16);
}
__device__ __forceinline__ float sigmoidf_(float x) { return 1.0f / (1.0f + __expf(-x)); }
__device__ __forceinline__ float siluf_(float x)    { return x * sigmoidf_(x); }

// ---------------- model constants ----------------
#define Bb 2
#define Tt 1024
#define Dd 2048
#define HK 16
#define HV 32
#define DK 128
#define DV 128
#define KEY_DIM 2048
#define VAL_DIM 4096
#define CONV_DIM 8192
#define QKVZ_DIM 12288
#define INTER 8192
#define BT (Bb*Tt)

// ================= fused residual add + gemma RMSNorm =================
__global__ void fused_add_rmsnorm(const float* __restrict__ x, const float* __restrict__ r,
                                  const float* __restrict__ w,
                                  float* __restrict__ res_out, float* __restrict__ h_out) {
  const int row = blockIdx.x;
  const float* xr = x + (size_t)row * Dd;
  const float* rr = r + (size_t)row * Dd;
  float* ro = res_out + (size_t)row * Dd;
  float* ho = h_out  + (size_t)row * Dd;
  float ss = 0.f;
  for (int i = threadIdx.x; i < Dd; i += blockDim.x) {
    float v = xr[i] + rr[i];
    ro[i] = v;
    ss += v * v;
  }
  #pragma unroll
  for (int m = 16; m; m >>= 1) ss += __shfl_xor(ss, m, 32);
  __shared__ float sred[8];
  const int wid = threadIdx.x >> 5, lid = threadIdx.x & 31;
  if (lid == 0) sred[wid] = ss;
  __syncthreads();
  __shared__ float sscale;
  if (threadIdx.x == 0) {
    float t = 0.f;
    #pragma unroll
    for (int i = 0; i < 8; ++i) t += sred[i];
    sscale = rsqrtf(t / (float)Dd + 1e-6f);
  }
  __syncthreads();
  const float scale = sscale;
  for (int i = threadIdx.x; i < Dd; i += blockDim.x)
    ho[i] = ro[i] * scale * (1.0f + w[i]);
}

#define LSTR 34   // even padded LDS stride (shorts) -> dword-aligned K-pair reads

// ================= large bf16 WMMA GEMM: 128x128 tile, 8 waves, double-buffered LDS =========
// C[MxN] = A[MxK] * B[KxN], row-major f32 in/out; M%128==0, N%128==0, K%32==0
#define GM 128
#define GN 128
#define GK 32

__global__ __launch_bounds__(256) void gemm_bf16_wmma128(
    const float* __restrict__ A, const float* __restrict__ B, float* __restrict__ C,
    int M, int N, int K) {
  __shared__ unsigned short sA[2][GM * LSTR];
  __shared__ unsigned short sB[2][GN * LSTR];   // transposed: sB[n*LSTR + k]

  const int tile_n = blockIdx.x * GN;
  const int tile_m = blockIdx.y * GM;
  const int tid  = threadIdx.x;
  const int wave = tid >> 5;
  const int lane = tid & 31;
  const int g    = lane >> 4;
  const int ml   = lane & 15;
  const int wr   = wave >> 1;     // 0..3 -> 32-row band
  const int wc   = wave & 1;      // 0..1 -> 64-col band

  v8f acc[2][4];
  #pragma unroll
  for (int i = 0; i < 2; ++i)
    #pragma unroll
    for (int nb = 0; nb < 4; ++nb) acc[i][nb] = (v8f){};

  float4 ra[4], rb[4];

  // ---- register-load one K-tile (global_load_b128) ----
  auto loadRegs = [&](int k0) {
    #pragma unroll
    for (int j = 0; j < 4; ++j) {
      int e = tid + j * 256;
      ra[j] = *(const float4*)&A[(size_t)(tile_m + (e >> 3)) * K + k0 + ((e & 7) << 2)];
      rb[j] = *(const float4*)&B[(size_t)(k0 + (e >> 5)) * N + tile_n + ((e & 31) << 2)];
    }
  };
  // ---- convert + store regs into LDS buffer p ----
  auto stageLds = [&](int p) {
    #pragma unroll
    for (int j = 0; j < 4; ++j) {
      int e = tid + j * 256;
      int r = e >> 3, c4 = (e & 7) << 2;
      unsigned int* pA = (unsigned int*)&sA[p][r * LSTR + c4];
      pA[0] = pack2bf(ra[j].x, ra[j].y);
      pA[1] = pack2bf(ra[j].z, ra[j].w);
      int kk = e >> 5, n4 = (e & 31) << 2;
      sB[p][(n4 + 0) * LSTR + kk] = f2bf(rb[j].x);
      sB[p][(n4 + 1) * LSTR + kk] = f2bf(rb[j].y);
      sB[p][(n4 + 2) * LSTR + kk] = f2bf(rb[j].z);
      sB[p][(n4 + 3) * LSTR + kk] = f2bf(rb[j].w);
    }
  };
  // ---- 8 WMMAs from LDS buffer p ----
  auto compute = [&](int p) {
    Frag a0, a1;
    #pragma unroll
    for (int v = 0; v < 8; ++v) {
      int kb = ((v & 4) << 2) + g * 8 + ((v & 3) << 1);  // (v>=4?16:0)+g*8+2*(v&3)
      a0.u[v] = *(const unsigned int*)&sA[p][(wr * 32 + ml) * LSTR + kb];
      a1.u[v] = *(const unsigned int*)&sA[p][(wr * 32 + 16 + ml) * LSTR + kb];
    }
    #pragma unroll
    for (int nb = 0; nb < 4; ++nb) {
      Frag bf;
      #pragma unroll
      for (int v = 0; v < 8; ++v) {
        int kk = ((v & 4) << 2) + g * 8 + ((v & 3) << 1);
        bf.u[v] = *(const unsigned int*)&sB[p][(wc * 64 + nb * 16 + ml) * LSTR + kk];
      }
      acc[0][nb] = __builtin_amdgcn_wmma_f32_16x16x32_bf16(false, a0.b, false, bf.b, (short)0, acc[0][nb], false, false);
      acc[1][nb] = __builtin_amdgcn_wmma_f32_16x16x32_bf16(false, a1.b, false, bf.b, (short)0, acc[1][nb], false, false);
    }
  };

  loadRegs(0);
  stageLds(0);
  __syncthreads();
  int p = 0;
  for (int k0 = 0; k0 < K; k0 += GK) {
    const bool has_next = (k0 + GK) < K;
    if (has_next) loadRegs(k0 + GK);   // global b128 loads in flight during compute
    compute(p);
    if (has_next) stageLds(p ^ 1);     // disjoint buffer: no WAR with current reads
    __syncthreads();
    p ^= 1;
  }

  #pragma unroll
  for (int i = 0; i < 2; ++i)
    #pragma unroll
    for (int nb = 0; nb < 4; ++nb) {
      #pragma unroll
      for (int v = 0; v < 8; ++v) {
        int row = tile_m + wr * 32 + i * 16 + g * 8 + v;
        C[(size_t)row * N + tile_n + wc * 64 + nb * 16 + ml] = acc[i][nb][v];
      }
    }
}

// ================= small bf16 WMMA GEMM: 64x64 tile (for N==64) =================
#define BM 64
#define BN 64
#define BK 32

__global__ __launch_bounds__(128) void gemm_bf16_wmma64(
    const float* __restrict__ A, const float* __restrict__ B, float* __restrict__ C,
    int M, int N, int K) {
  __shared__ unsigned short sA[BM * LSTR];
  __shared__ unsigned short sB[BN * LSTR];

  const int tile_n = blockIdx.x * BN;
  const int tile_m = blockIdx.y * BM;
  const int tid  = threadIdx.x;
  const int wave = tid >> 5;
  const int lane = tid & 31;
  const int g    = lane >> 4;
  const int ml   = lane & 15;

  v8f acc0 = {}, acc1 = {}, acc2 = {}, acc3 = {};
  const int arow = wave * 16 + ml;

  for (int k0 = 0; k0 < K; k0 += BK) {
    #pragma unroll
    for (int j = 0; j < (BM * BK) / 128; ++j) {
      int e  = tid + j * 128;
      int rA = e >> 5, cA = e & 31;
      sA[rA * LSTR + cA] = f2bf(A[(size_t)(tile_m + rA) * K + k0 + cA]);
      int kB = e >> 6, nB = e & 63;
      sB[nB * LSTR + kB] = f2bf(B[(size_t)(k0 + kB) * N + tile_n + nB]);
    }
    if (k0 + BK < K) {   // global_prefetch_b8 for next K-tile
      __builtin_prefetch(&A[(size_t)(tile_m + (tid & 63)) * K + k0 + BK], 0, 1);
      __builtin_prefetch(&B[(size_t)(k0 + BK + (tid & 31)) * N + tile_n], 0, 1);
    }
    __syncthreads();

    Frag af;
    #pragma unroll
    for (int v = 0; v < 8; ++v) {
      int kb = ((v & 4) << 2) + g * 8 + ((v & 3) << 1);
      af.u[v] = *(const unsigned int*)&sA[arow * LSTR + kb];
    }
    Frag bf0, bf1, bf2, bf3;
    #pragma unroll
    for (int v = 0; v < 8; ++v) {
      int kk = ((v & 4) << 2) + g * 8 + ((v & 3) << 1);
      bf0.u[v] = *(const unsigned int*)&sB[(0 * 16 + ml) * LSTR + kk];
      bf1.u[v] = *(const unsigned int*)&sB[(1 * 16 + ml) * LSTR + kk];
      bf2.u[v] = *(const unsigned int*)&sB[(2 * 16 + ml) * LSTR + kk];
      bf3.u[v] = *(const unsigned int*)&sB[(3 * 16 + ml) * LSTR + kk];
    }
    acc0 = __builtin_amdgcn_wmma_f32_16x16x32_bf16(false, af.b, false, bf0.b, (short)0, acc0, false, false);
    acc1 = __builtin_amdgcn_wmma_f32_16x16x32_bf16(false, af.b, false, bf1.b, (short)0, acc1, false, false);
    acc2 = __builtin_amdgcn_wmma_f32_16x16x32_bf16(false, af.b, false, bf2.b, (short)0, acc2, false, false);
    acc3 = __builtin_amdgcn_wmma_f32_16x16x32_bf16(false, af.b, false, bf3.b, (short)0, acc3, false, false);
    __syncthreads();
  }

  #pragma unroll
  for (int v = 0; v < 8; ++v) {
    int rowo = tile_m + wave * 16 + g * 8 + v;
    float* Crow = C + (size_t)rowo * N + tile_n;
    Crow[0 * 16 + ml] = acc0[v];
    Crow[1 * 16 + ml] = acc1[v];
    Crow[2 * 16 + ml] = acc2[v];
    Crow[3 * 16 + ml] = acc3[v];
  }
}

// ================= causal depthwise conv (K=4) + SiLU over first 8192 qkvz channels ==========
__global__ void conv_silu(const float* __restrict__ qkvz, const float* __restrict__ w,
                          float* __restrict__ out) {
  const int idx = blockIdx.x * blockDim.x + threadIdx.x;   // B*T*CONV_DIM
  const int c  = idx & (CONV_DIM - 1);
  const int bt = idx >> 13;
  const int t  = bt & (Tt - 1);
  const int b  = bt >> 10;
  float acc = 0.f;
  #pragma unroll
  for (int i = 0; i < 4; ++i) {
    int tt = t - 3 + i;
    if (tt >= 0)
      acc += qkvz[((size_t)(b * Tt + tt)) * QKVZ_DIM + c] * w[c * 4 + i];
  }
  out[idx] = siluf_(acc);
}

// ================= l2-normalize q and k heads (warp per 128-elem row) =================
__global__ void l2norm_qk(const float* __restrict__ conv, float* __restrict__ qs,
                          float* __restrict__ ks) {
  const int warp = (blockIdx.x * blockDim.x + threadIdx.x) >> 5;
  const int lane = threadIdx.x & 31;
  const int NR = BT * HK;
  if (warp >= 2 * NR) return;
  const int isq = (warp < NR) ? 1 : 0;
  const int r   = isq ? warp : warp - NR;
  const int bt  = r >> 4, hk = r & 15;
  const float* src = conv + (size_t)bt * CONV_DIM + (isq ? 0 : KEY_DIM) + hk * DK;
  float vals[4]; float ss = 0.f;
  #pragma unroll
  for (int j = 0; j < 4; ++j) { vals[j] = src[lane + j * 32]; ss += vals[j] * vals[j]; }
  #pragma unroll
  for (int m = 16; m; m >>= 1) ss += __shfl_xor(ss, m, 32);
  float scale = rsqrtf(ss + 1e-6f);
  if (isq) scale *= 0.08838834764831845f;   // DK^-0.5
  float* dst = (isq ? qs : ks) + (size_t)r * DK;
  #pragma unroll
  for (int j = 0; j < 4; ++j) dst[lane + j * 32] = vals[j] * scale;
}

// ================= beta / decay =================
__global__ void beta_decay(const float* __restrict__ ba, const float* __restrict__ A_log,
                           const float* __restrict__ dt_bias,
                           float* __restrict__ beta, float* __restrict__ decay) {
  const int idx = blockIdx.x * blockDim.x + threadIdx.x;
  const int h = idx & 31, bt = idx >> 5;
  const float bb = ba[(size_t)bt * 64 + h];
  const float aa = ba[(size_t)bt * 64 + 32 + h];
  beta[idx] = sigmoidf_(bb);
  float x  = aa + dt_bias[h];
  float sp = (x > 20.f) ? x : log1pf(__expf(x));
  decay[idx] = __expf(-__expf(A_log[h]) * sp);
}

// ================= delta-rule recurrent scan: one block per (b, head), state in VGPRs ========
__global__ __launch_bounds__(128) void delta_scan(
    const float* __restrict__ qs, const float* __restrict__ ks,
    const float* __restrict__ conv, const float* __restrict__ beta,
    const float* __restrict__ decay, float* __restrict__ o) {
  const int b  = blockIdx.x >> 5;
  const int h  = blockIdx.x & 31;
  const int hk = h >> 1;
  const int v  = threadIdx.x;
  __shared__ float sK[DK], sQ[DK];
  float S[DK];
  #pragma unroll
  for (int i = 0; i < DK; ++i) S[i] = 0.f;

  for (int t = 0; t < Tt; ++t) {
    const size_t bt = (size_t)b * Tt + t;
    sK[v] = ks[(bt * HK + hk) * DK + v];
    sQ[v] = qs[(bt * HK + hk) * DK + v];
    __syncthreads();
    const float vt  = conv[bt * CONV_DIM + 2 * KEY_DIM + h * DV + v];
    const float bta = beta[bt * HV + h];
    const float dec = decay[bt * HV + h];
    float vpred = 0.f;
    #pragma unroll
    for (int i = 0; i < DK; ++i) { S[i] *= dec; vpred += sK[i] * S[i]; }
    const float delta = (vt - vpred) * bta;
    float oo = 0.f;
    #pragma unroll
    for (int i = 0; i < DK; ++i) { S[i] += sK[i] * delta; oo += sQ[i] * S[i]; }
    o[(bt * HV + h) * DV + v] = oo;
    __syncthreads();
  }
}

// ================= gated per-head RMSNorm (in-place on o) =================
__global__ void gated_norm(float* __restrict__ o, const float* __restrict__ qkvz,
                           const float* __restrict__ nw) {
  const int warp = (blockIdx.x * blockDim.x + threadIdx.x) >> 5;
  const int lane = threadIdx.x & 31;
  const int h = warp & 31, bt = warp >> 5;
  float vals[4]; float ss = 0.f;
  #pragma unroll
  for (int j = 0; j < 4; ++j) {
    int dv = lane + j * 32;
    float z = qkvz[(size_t)bt * QKVZ_DIM + CONV_DIM + h * DV + dv];
    float x = o[(size_t)warp * DV + dv] * siluf_(z);
    vals[j] = x; ss += x * x;
  }
  #pragma unroll
  for (int m = 16; m; m >>= 1) ss += __shfl_xor(ss, m, 32);
  const float inv = rsqrtf(ss / (float)DV + 1e-6f);
  #pragma unroll
  for (int j = 0; j < 4; ++j) {
    int dv = lane + j * 32;
    o[(size_t)warp * DV + dv] = vals[j] * inv * nw[dv];
  }
}

// ================= SiLU(g) * u, in place on g =================
__global__ void silu_mul(float* __restrict__ g, const float* __restrict__ u, int n) {
  const int i = blockIdx.x * blockDim.x + threadIdx.x;
  if (i < n) g[i] = siluf_(g[i]) * u[i];
}

// ================= host launcher =================
extern "C" void kernel_launch(void* const* d_in, const int* in_sizes, int n_in,
                              void* d_out, int out_size, void* d_ws, size_t ws_size,
                              hipStream_t stream) {
  (void)in_sizes; (void)n_in; (void)out_size; (void)ws_size;
  const float* hidden   = (const float*)d_in[1];
  const float* residual = (const float*)d_in[2];
  const float* ln_in_w  = (const float*)d_in[3];
  const float* ln_post  = (const float*)d_in[4];
  const float* Wqkvz    = (const float*)d_in[5];
  const float* Wba      = (const float*)d_in[6];
  const float* conv_w   = (const float*)d_in[7];
  const float* A_log    = (const float*)d_in[8];
  const float* dt_bias  = (const float*)d_in[9];
  const float* norm_w   = (const float*)d_in[10];
  const float* Wout     = (const float*)d_in[11];
  const float* Wgate    = (const float*)d_in[12];
  const float* Wup      = (const float*)d_in[13];
  const float* Wdown    = (const float*)d_in[14];
  float* out = (float*)d_out;

  const size_t BTD = (size_t)BT * Dd;
  float* ws = (float*)d_ws;
  float* res1  = ws;
  float* h     = res1  + BTD;                    // reused as h2
  float* qkvz  = h     + BTD;                    // reused as g
  float* ba    = qkvz  + (size_t)BT * QKVZ_DIM;
  float* convb = ba    + (size_t)BT * 64;        // reused as u
  float* qsb   = convb + (size_t)BT * CONV_DIM;
  float* ksb   = qsb   + BTD;
  float* betab = ksb   + BTD;
  float* decb  = betab + (size_t)BT * HV;
  float* ob    = decb  + (size_t)BT * HV;        // gated norm in-place
  float* attn  = ob    + (size_t)BT * VAL_DIM;
  float* gbuf  = qkvz;
  float* ubuf  = convb;
  float* res2  = out + BTD;

  // 1) res1 = hidden + residual ; h = rmsnorm(res1)
  fused_add_rmsnorm<<<BT, 256, 0, stream>>>(hidden, residual, ln_in_w, res1, h);
  // 2) qkvz = h @ Wqkvz ; ba = h @ Wba
  gemm_bf16_wmma128<<<dim3(QKVZ_DIM / GN, BT / GM), 256, 0, stream>>>(h, Wqkvz, qkvz, BT, QKVZ_DIM, Dd);
  gemm_bf16_wmma64<<<dim3(1, BT / BM), 128, 0, stream>>>(h, Wba, ba, BT, 64, Dd);
  // 3) causal conv + SiLU ; l2norm(q,k) ; beta/decay
  conv_silu<<<(BT * CONV_DIM) / 256, 256, 0, stream>>>(qkvz, conv_w, convb);
  l2norm_qk<<<(2 * BT * HK * 32) / 256, 256, 0, stream>>>(convb, qsb, ksb);
  beta_decay<<<(BT * HV) / 256, 256, 0, stream>>>(ba, A_log, dt_bias, betab, decb);
  // 4) recurrent delta-rule scan
  delta_scan<<<Bb * HV, DK, 0, stream>>>(qsb, ksb, convb, betab, decb, ob);
  // 5) gated per-head RMSNorm (in place on o)
  gated_norm<<<(BT * HV * 32) / 256, 256, 0, stream>>>(ob, qkvz, norm_w);
  // 6) attn = o @ Wout ; res2 = attn + res1 ; h2 = rmsnorm(res2)
  gemm_bf16_wmma128<<<dim3(Dd / GN, BT / GM), 256, 0, stream>>>(ob, Wout, attn, BT, Dd, VAL_DIM);
  fused_add_rmsnorm<<<BT, 256, 0, stream>>>(attn, res1, ln_post, res2, h);
  // 7) MLP: g = h2@Wgate ; u = h2@Wup ; g = silu(g)*u ; out = g@Wdown
  gemm_bf16_wmma128<<<dim3(INTER / GN, BT / GM), 256, 0, stream>>>(h, Wgate, gbuf, BT, INTER, Dd);
  gemm_bf16_wmma128<<<dim3(INTER / GN, BT / GM), 256, 0, stream>>>(h, Wup, ubuf, BT, INTER, Dd);
  silu_mul<<<(BT * INTER) / 256, 256, 0, stream>>>(gbuf, ubuf, BT * INTER);
  gemm_bf16_wmma128<<<dim3(Dd / GN, BT / GM), 256, 0, stream>>>(gbuf, Wdown, out, BT, Dd, INTER);
}
